// ConvKNRM_47785806135495
// MI455X (gfx1250) — matrix operand order repaired and checked by reading, not compiled, not run on hardware
//
#include <hip/hip_runtime.h>
#include <hip/hip_bf16.h>
#include <cstdint>
#include <cstddef>

typedef __attribute__((ext_vector_type(16))) _Float16 v16h;
typedef __attribute__((ext_vector_type(8)))  _Float16 v8h;
typedef __attribute__((ext_vector_type(8)))  float    v8f;

#define B_     128
#define LQ_    64
#define LD_    2048
#define EMB_   100
#define NF_    64
#define NG_    3
#define NK_    11
#define VOCAB_ 50000

// mu_i = 0.1 + 0.2*i - 1 for i<10; i==10 -> mu=1.0, sigma=0.001 (exact bin)
__constant__ float c_mu[NK_]  = {-0.9f,-0.7f,-0.5f,-0.3f,-0.1f,0.1f,0.3f,0.5f,0.7f,0.9f,1.0f};
// 0.5/sigma^2
__constant__ float c_inv[NK_] = {50.f,50.f,50.f,50.f,50.f,50.f,50.f,50.f,50.f,50.f,500000.f};

// ---------------------------------------------------------------------------
// Kernel 0: pre-swizzle conv weights into exact WMMA B-fragment layout (f16).
// Wt[ny][kb][lane][h] with ny = slice*4 + ftile (24), kb = k-step (4),
// value = B(lane,h) = w_ng[fb + (lane&15), e, j], e = kb*32 + (h&7) + (h>=8?16:0) + (lane>=16?8:0)
// (zero-padded for e >= 100). 49152 halves = 96 KB; fully L2-resident.
// ---------------------------------------------------------------------------
__global__ void wprep_kernel(const float* __restrict__ w1,
                             const float* __restrict__ w2,
                             const float* __restrict__ w3,
                             _Float16* __restrict__ Wt)
{
    int idx  = blockIdx.x * blockDim.x + threadIdx.x;   // < 24*4*32*16
    int h    = idx & 15;
    int lane = (idx >> 4) & 31;
    int kb   = (idx >> 9) & 3;
    int ny   = idx >> 11;                               // 0..23
    int slice = ny >> 2;
    int fb    = (ny & 3) << 4;

    int ng, j;
    const float* w;
    switch (slice) {
        case 0:  ng = 1; j = 0; w = w1; break;
        case 1:  ng = 2; j = 0; w = w2; break;
        case 2:  ng = 2; j = 1; w = w2; break;
        case 3:  ng = 3; j = 0; w = w3; break;
        case 4:  ng = 3; j = 1; w = w3; break;
        default: ng = 3; j = 2; w = w3; break;
    }
    int fg = fb + (lane & 15);
    int e  = kb * 32 + (h & 7) + ((h >> 3) << 4) + ((lane >= 16) ? 8 : 0);
    float v = (e < EMB_) ? w[(size_t)fg * EMB_ * ng + (size_t)e * ng + j] : 0.f;
    Wt[idx] = (_Float16)v;
}

// ---------------------------------------------------------------------------
// Kernel 1: project embedding table through all 6 conv weight slices.
// P[slice][v][f] = sum_e emb[v,e] * w_ng[f,e,j]   (raw, pre-bias/tanh)
// One block (8 waves) per 16-row vocab tile: A staged once in LDS (f16,
// K padded to 128), each wave serves 3 of the 24 N-tiles with
// 4 x v_wmma_f32_16x16x32_f16. B fragments come straight from Wt as two
// aligned 16B vector loads per k-step.
// ---------------------------------------------------------------------------
__global__ void proj_kernel(const float* __restrict__ emb,
                            const _Float16* __restrict__ Wt,
                            _Float16* __restrict__ P)
{
    __shared__ _Float16 As[16][128];
    const int vt  = blockIdx.x;                          // vocab tile 0..3124
    const int tid = threadIdx.x;

    // Cooperative coalesced load + f16 convert of the A tile (zero-pad K).
    for (int idx = tid; idx < 16 * 128; idx += 256) {
        int row = idx >> 7;
        int e   = idx & 127;
        As[row][e] = (e < EMB_) ? (_Float16)emb[(size_t)(vt * 16 + row) * EMB_ + e]
                                : (_Float16)0.f;
    }
    __syncthreads();

    const int wave = tid >> 5;
    const int lane = tid & 31;
    const int l15  = lane & 15;
    const int hi   = lane >> 4;

    // A fragments for all 4 k-steps, read once from LDS (ds_load_b128 pairs).
    // A layout (16-bit 16x32): K = h + (h>=8 ? 8 : 0) + (lane>=16 ? 8 : 0)
    v16h A[4];
    #pragma unroll
    for (int kb = 0; kb < 4; ++kb) {
        const _Float16* base = &As[l15][kb * 32 + (hi ? 8 : 0)];
        v8h lo = *(const v8h*)(base);
        v8h hb = *(const v8h*)(base + 16);
        #pragma unroll
        for (int i = 0; i < 8; ++i) { A[kb][i] = lo[i]; A[kb][8 + i] = hb[i]; }
    }

    #pragma unroll
    for (int t = 0; t < 3; ++t) {
        const int ny    = wave * 3 + t;                  // 0..23
        const int slice = ny >> 2;
        const int fb    = (ny & 3) << 4;

        v8f c = {};
        #pragma unroll
        for (int kb = 0; kb < 4; ++kb) {
            const _Float16* bbase = Wt + (((size_t)ny * 4 + kb) * 32 + lane) * 16;
            v8h lo = *(const v8h*)(bbase);
            v8h hb = *(const v8h*)(bbase + 8);
            v16h Bm;
            #pragma unroll
            for (int i = 0; i < 8; ++i) { Bm[i] = lo[i]; Bm[8 + i] = hb[i]; }
            c = __builtin_amdgcn_wmma_f32_16x16x32_f16(false, A[kb], false, Bm,
                                                       (short)0, c, false, false);
        }
        // D layout: row M = r + (lane>=16 ? 8 : 0), col N = lane&15
        _Float16* Ps = P + (size_t)slice * VOCAB_ * NF_;
        const int v0 = vt * 16 + (hi ? 8 : 0);
        #pragma unroll
        for (int r = 0; r < 8; ++r)
            Ps[(size_t)(v0 + r) * NF_ + fb + l15] = (_Float16)c[r];
    }
}

// ---------------------------------------------------------------------------
// Kernel 2: query conv via gathered projections + bias + tanh -> Qc (f16)
// Qc[ng][b][q][f]  (row-major M=q, K=f : A-operand storage for sim GEMM)
// ---------------------------------------------------------------------------
__global__ void qgather_kernel(const long long* __restrict__ qtok,
                               const _Float16* __restrict__ P,
                               const float* __restrict__ b1,
                               const float* __restrict__ b2,
                               const float* __restrict__ b3,
                               _Float16* __restrict__ Qc)
{
    int idx = blockIdx.x * blockDim.x + threadIdx.x;   // < 3*128*64*64
    int f  = idx & 63;
    int q  = (idx >> 6) & 63;
    int b  = (idx >> 12) & 127;
    int ng = idx >> 19;                                // 0..2

    const long long* qb = qtok + (size_t)b * LQ_;
    const float* bias = (ng == 0) ? b1 : (ng == 1 ? b2 : b3);
    const int sbase   = (ng == 0) ? 0  : (ng == 1 ? 1  : 3);

    float acc = bias[f];
    for (int j = 0; j <= ng; ++j) {
        int t = q + j;
        if (t < LQ_) {                                  // right zero-pad: skip
            int tok = (int)qb[t];
            acc += (float)P[((size_t)(sbase + j) * VOCAB_ + tok) * NF_ + f];
        }
    }
    Qc[(((size_t)ng * B_ + b) * LQ_ + q) * NF_ + f] = (_Float16)tanhf(acc);
}

// ---------------------------------------------------------------------------
// Kernel 3: doc conv via gathered projections -> Dc (f16), one ngram at a time
// Dc[b][t][f]  (f contiguous : B-operand storage, K=f, N=t)
// ---------------------------------------------------------------------------
__global__ void dgather_kernel(const long long* __restrict__ dtok,
                               const _Float16* __restrict__ P,
                               const float* __restrict__ bias,
                               _Float16* __restrict__ Dc,
                               int ng, int sbase)
{
    int idx = blockIdx.x * blockDim.x + threadIdx.x;   // < 128*2048*16
    int fg = (idx & 15) << 2;                          // group of 4 filters
    int t  = (idx >> 4) & 2047;
    int b  = idx >> 15;

    const long long* db = dtok + (size_t)b * LD_;
    float a0 = bias[fg], a1 = bias[fg + 1], a2 = bias[fg + 2], a3 = bias[fg + 3];
    for (int j = 0; j < ng; ++j) {
        int tt = t + j;
        if (tt < LD_) {
            int tok = (int)db[tt];
            const _Float16* p = P + ((size_t)(sbase + j) * VOCAB_ + tok) * NF_ + fg;
            a0 += (float)p[0]; a1 += (float)p[1]; a2 += (float)p[2]; a3 += (float)p[3];
        }
    }
    _Float16* o = Dc + ((size_t)b * LD_ + t) * NF_ + fg;
    o[0] = (_Float16)tanhf(a0); o[1] = (_Float16)tanhf(a1);
    o[2] = (_Float16)tanhf(a2); o[3] = (_Float16)tanhf(a3);
}

// ---------------------------------------------------------------------------
// Kernel 4: fused sim-GEMM + RBF soft-bin + d-sum + log1p + q-sum.
// Block = (b, q-tile of 16) = 256 threads = 8 waves. Each wave sweeps
// d-tiles dt = wave, wave+8, ... (128 tiles). Per tile: two f16 WMMAs
// (K=64), then 11 exps per f32 accumulator element, fused — the sim
// tensor is never written to memory. Per-lane acc[8 rows][11 kernels].
// ---------------------------------------------------------------------------
__global__ void sim_kernel(const _Float16* __restrict__ Qc,
                           const _Float16* __restrict__ Dc,
                           float* __restrict__ counts,
                           int ng_idx)
{
    __shared__ float qt[16][NK_];

    const int b     = blockIdx.x >> 2;
    const int qtile = blockIdx.x & 3;
    const int tid   = threadIdx.x;
    const int wave  = tid >> 5;
    const int lane  = tid & 31;
    const int l15   = lane & 15;
    const int hi    = lane >> 4;

    if (tid < 16 * NK_) ((float*)qt)[tid] = 0.f;
    __syncthreads();

    // Load the wave's A tiles once (16 q-rows x 64 f, two k-steps of 32).
    const _Float16* Qb = Qc + (((size_t)ng_idx * B_ + b) * LQ_ + qtile * 16) * NF_;
    v16h A[2];
    #pragma unroll
    for (int kb = 0; kb < 2; ++kb) {
        const _Float16* base = Qb + (size_t)l15 * NF_ + kb * 32 + (hi ? 8 : 0);
        v8h lo  = *(const v8h*)(base);        // K chunk [off, off+8)
        v8h hi8 = *(const v8h*)(base + 16);   // K chunk [off+16, off+24)
        #pragma unroll
        for (int i = 0; i < 8; ++i) { A[kb][i] = lo[i]; A[kb][8 + i] = hi8[i]; }
    }

    float acc[8][NK_];
    #pragma unroll
    for (int r = 0; r < 8; ++r)
        #pragma unroll
        for (int k = 0; k < NK_; ++k) acc[r][k] = 0.f;

    const _Float16* Db = Dc + (size_t)b * LD_ * NF_;
    for (int dt = wave; dt < LD_ / 16; dt += 8) {       // 16 uniform iters/wave
        if (dt + 8 < LD_ / 16)                           // wave-uniform guard
            __builtin_prefetch(Db + (size_t)(dt + 8) * 16 * NF_, 0, 0);

        v8f c = {};
        #pragma unroll
        for (int kb = 0; kb < 2; ++kb) {
            const int dpos = dt * 16 + l15;             // B col N = lane&15
            const _Float16* base = Db + (size_t)dpos * NF_ + kb * 32 + (hi ? 8 : 0);
            v8h lo  = *(const v8h*)(base);              // K chunk [off, off+8)
            v8h hi8 = *(const v8h*)(base + 16);         // K chunk [off+16, off+24)
            v16h Bm;
            #pragma unroll
            for (int i = 0; i < 8; ++i) { Bm[i] = lo[i]; Bm[8 + i] = hi8[i]; }
            c = __builtin_amdgcn_wmma_f32_16x16x32_f16(false, A[kb], false, Bm,
                                                       (short)0, c, false, false);
        }
        // Fused RBF soft-bin accumulate on the f32 accumulator tile.
        #pragma unroll
        for (int r = 0; r < 8; ++r) {
            float s = c[r];
            #pragma unroll
            for (int k = 0; k < NK_; ++k) {
                float dd = s - c_mu[k];
                acc[r][k] += __expf(-c_inv[k] * dd * dd);
            }
        }
    }

    // Cross-lane / cross-wave reduction of the d-sum into LDS.
    #pragma unroll
    for (int r = 0; r < 8; ++r) {
        int m = r + (hi ? 8 : 0);
        #pragma unroll
        for (int k = 0; k < NK_; ++k) atomicAdd(&qt[m][k], acc[r][k]);
    }
    __syncthreads();

    // log1p per (q-row, kernel), then accumulate over q into counts[b, col].
    if (tid < 16 * NK_) {
        int m = tid / NK_;
        int k = tid % NK_;
        float v = log1pf(qt[m][k]);
        atomicAdd(&counts[(size_t)b * (NK_ * NG_) + ng_idx * NK_ + k], v);
    }
}

// ---------------------------------------------------------------------------
// Kernel 5: linear head  score[b] = counts[b,:] . lin_w
// ---------------------------------------------------------------------------
__global__ void final_kernel(const float* __restrict__ counts,
                             const float* __restrict__ lin_w,
                             float* __restrict__ out)
{
    int b = threadIdx.x;
    if (b < B_) {
        float s = 0.f;
        #pragma unroll
        for (int j = 0; j < NK_ * NG_; ++j)
            s += counts[(size_t)b * (NK_ * NG_) + j] * lin_w[j];
        out[b] = s;
    }
}

// ---------------------------------------------------------------------------
extern "C" void kernel_launch(void* const* d_in, const int* in_sizes, int n_in,
                              void* d_out, int out_size, void* d_ws, size_t ws_size,
                              hipStream_t stream)
{
    (void)in_sizes; (void)n_in; (void)out_size; (void)ws_size;

    const long long* q   = (const long long*)d_in[0];   // int64 tokens
    const long long* dtk = (const long long*)d_in[1];   // int64 tokens
    // d_in[2]=qlen, d_in[3]=dlen : unused by the reference
    const float* emb   = (const float*)d_in[4];
    const float* lin_w = (const float*)d_in[5];
    const float* w1 = (const float*)d_in[6];  const float* b1 = (const float*)d_in[7];
    const float* w2 = (const float*)d_in[8];  const float* b2 = (const float*)d_in[9];
    const float* w3 = (const float*)d_in[10]; const float* b3 = (const float*)d_in[11];

    // Workspace layout (all 256B aligned):
    //   P      @ 0          : 6 * 50000 * 64 f16 = 38,400,000 B
    //   Qc     @ 38,400,000 : 3 * 128 * 64 * 64 f16 = 3,145,728 B
    //   Dc     @ 41,545,728 : 128 * 2048 * 64 f16 = 33,554,432 B (reused/ngram)
    //   counts @ 75,100,160 : 128 * 33 f32 = 16,896 B
    //   Wt     @ 75,133,952 : 24*4*32*16 f16 = 98,304 B (B-fragment weights)
    char* ws = (char*)d_ws;
    _Float16* P      = (_Float16*)(ws);
    _Float16* Qc     = (_Float16*)(ws + 38400000);
    _Float16* Dc     = (_Float16*)(ws + 38400000 + 3145728);
    float*    counts = (float*)   (ws + 75100160);
    _Float16* Wt     = (_Float16*)(ws + 75133952);

    hipMemsetAsync(counts, 0, B_ * NK_ * NG_ * sizeof(float), stream);

    wprep_kernel<<<(24 * 4 * 32 * 16) / 256, 256, 0, stream>>>(w1, w2, w3, Wt);

    proj_kernel<<<VOCAB_ / 16, 256, 0, stream>>>(emb, Wt, P);

    qgather_kernel<<<(NG_ * B_ * LQ_ * NF_) / 256, 256, 0, stream>>>(q, P, b1, b2, b3, Qc);

    const float* biases[NG_] = {b1, b2, b3};
    const int    sbase[NG_]  = {0, 1, 3};
    for (int ng = 0; ng < NG_; ++ng) {
        dgather_kernel<<<(B_ * LD_ * 16) / 256, 256, 0, stream>>>(
            dtk, P, biases[ng], Dc, ng + 1, sbase[ng]);
        sim_kernel<<<B_ * 4, 256, 0, stream>>>(Qc, Dc, counts, ng);
    }

    final_kernel<<<1, 128, 0, stream>>>(counts, lin_w, (float*)d_out);
}
